// SqueezeNet_38414187495943
// MI455X (gfx1250) — matrix operand order repaired and checked by reading, not compiled, not run on hardware
//
#include <hip/hip_runtime.h>
#include <hip/hip_bf16.h>

typedef _Float16 h16;
typedef __attribute__((ext_vector_type(16))) _Float16 v16h;
typedef __attribute__((ext_vector_type(8)))  _Float16 v8h;
typedef __attribute__((ext_vector_type(8)))  float    v8f;

union AFrag { v16h v; v8h h[2]; };

#define WMMA_F16(A, B, C) __builtin_amdgcn_wmma_f32_16x16x32_f16( \
    false, (A), false, (B), (short)0, (C), false, false)

// Invariant used throughout: M = 32*H*W is a multiple of 128 for every layer
// in this network (100352 / 25088 / 6272), and grids are launched exactly, so
// every wave's 16 M-rows are valid -- no M bounds checks needed anywhere.

// ---------------- prep kernels ----------------

// Transpose HWIO weight [K x Cout] -> [Cout x Kpad] (K-padded with zeros),
// binarizing (sign, with sign(0)=0) for fire convs, plain f32->f16 for stem.
__global__ void prep_wT_kernel(const float* __restrict__ w, h16* __restrict__ o,
                               int K, int Kpad, int Cout, int bin) {
  long total = (long)Cout * Kpad;
  long i  = (long)blockIdx.x * blockDim.x + threadIdx.x;
  long st = (long)gridDim.x * blockDim.x;
  for (; i < total; i += st) {
    int k = (int)(i % Kpad);
    int n = (int)(i / Kpad);
    float v = 0.f;
    if (k < K) {
      float x = w[(size_t)k * Cout + n];
      v = bin ? ((x > 0.f ? 1.f : 0.f) - (x < 0.f ? 1.f : 0.f)) : x;
    }
    o[i] = (h16)v;
  }
}

__global__ void fold_bn_kernel(const float* __restrict__ g, const float* __restrict__ b,
                               const float* __restrict__ m, const float* __restrict__ v,
                               float* __restrict__ sc, float* __restrict__ sh, int c) {
  int i = blockIdx.x * blockDim.x + threadIdx.x;
  if (i < c) {
    float s = g[i] * rsqrtf(v[i] + 1e-3f);
    sc[i] = s;
    sh[i] = b[i] - m[i] * s;
  }
}

__global__ void cast_f32_h_kernel(const float* __restrict__ x, h16* __restrict__ o, long n) {
  long i  = (long)blockIdx.x * blockDim.x + threadIdx.x;
  long st = (long)gridDim.x * blockDim.x;
  for (; i < n; i += st) o[i] = (h16)x[i];
}

// ---------------- shared WMMA epilogue: relu -> folded BN -> f16 ----------------
// C/D layout: col = lane%16, rows = 8*(lane/16) + r. Two N tiles per wave.

__device__ __forceinline__ void bn_epilogue(
    const v8f& acc0, const v8f& acc1,
    const float* __restrict__ scale, const float* __restrict__ shift,
    const float* __restrict__ bias, h16* __restrict__ out,
    int mtile, int khalf, int lrow, int nbase, int CoutTot, int coutOff) {
  int n0 = nbase + lrow;
  int n1 = n0 + 16;
  float s0 = scale[n0], s1 = scale[n1];
  float t0 = shift[n0], t1 = shift[n1];
  float b0 = bias ? bias[n0] : 0.f;
  float b1 = bias ? bias[n1] : 0.f;
#pragma unroll
  for (int r = 0; r < 8; ++r) {
    int mr = mtile * 16 + 8 * khalf + r;
    float z0 = fmaxf(acc0[r] + b0, 0.f) * s0 + t0;
    float z1 = fmaxf(acc1[r] + b1, 0.f) * s1 + t1;
    h16* po = out + (size_t)mr * CoutTot + coutOff;
    po[n0] = (h16)z0;
    po[n1] = (h16)z1;
  }
}

// ---------------- 1x1 conv == GEMM (squeeze / expand1x1) ----------------
// Wave computes 16(M) x 32(N); one A fragment feeds two WMMAs per 32-K step.
// B is a direct 32B-aligned vector load from wT[Cout][K] (K = Cin, mult of 32).

__global__ __launch_bounds__(256) void gemm_bn_wmma(
    const h16* __restrict__ in, const h16* __restrict__ wT,
    const float* __restrict__ scale, const float* __restrict__ shift,
    h16* __restrict__ out, int K, int CoutTot, int coutOff)
{
  const int lane  = threadIdx.x & 31;
  const int wave  = threadIdx.x >> 5;
  const int lrow  = lane & 15;
  const int khalf = lane >> 4;
  const int mtile = blockIdx.x * 8 + wave;
  const int nbase = blockIdx.y * 32;
  const int m     = mtile * 16 + lrow;        // always < M (see invariant)

  const h16* pa = in + (size_t)m * K + 16 * khalf;           // 16B aligned
  const h16* w0 = wT + (size_t)(nbase + lrow) * K + 16 * khalf;
  const h16* w1 = w0 + (size_t)16 * K;

  v8f acc0 = {};
  v8f acc1 = {};
  for (int k0 = 0; k0 < K; k0 += 32) {
    AFrag a;
    a.h[0] = *(const v8h*)(pa + k0);
    a.h[1] = *(const v8h*)(pa + k0 + 8);
    __builtin_prefetch(pa + k0 + 64, 0, 3);
    v16h b0 = *(const v16h*)(w0 + k0);
    v16h b1 = *(const v16h*)(w1 + k0);
    acc0 = WMMA_F16(a.v, b0, acc0);
    acc1 = WMMA_F16(a.v, b1, acc1);
  }
  bn_epilogue(acc0, acc1, scale, shift, nullptr, out,
              mtile, khalf, lrow, nbase, CoutTot, coutOff);
}

// ---------------- 3x3 s1 SAME conv (expand3x3), Cin % 32 == 0 ----------------
// Unrolled 9-tap loop x channel-chunk loop: no integer division in the loop,
// each 32-K step sits inside one filter tap, SAME padding -> zero A fragment.

__global__ __launch_bounds__(256) void conv3_bn_wmma(
    const h16* __restrict__ in, const h16* __restrict__ wT,
    const float* __restrict__ scale, const float* __restrict__ shift,
    h16* __restrict__ out, int H, int W, int Cin, int CoutTot, int coutOff)
{
  const int lane  = threadIdx.x & 31;
  const int wave  = threadIdx.x >> 5;
  const int lrow  = lane & 15;
  const int khalf = lane >> 4;
  const int mtile = blockIdx.x * 8 + wave;
  const int nbase = blockIdx.y * 32;
  const int Kpad  = 9 * Cin;
  const int m     = mtile * 16 + lrow;        // always < M (see invariant)

  const int b  = m / (H * W);
  const int r0 = m % (H * W);
  const int oy = r0 / W;
  const int ox = r0 % W;

  const h16* w0 = wT + (size_t)(nbase + lrow) * Kpad + 16 * khalf;
  const h16* w1 = w0 + (size_t)16 * Kpad;

  v8f acc0 = {};
  v8f acc1 = {};
  v8h z = {};
#pragma unroll
  for (int t = 0; t < 9; ++t) {
    const int ky = t / 3, kx = t % 3;                 // compile-time constants
    const int iy = oy - 1 + ky;
    const int ix = ox - 1 + kx;
    const bool inb = (unsigned)iy < (unsigned)H && (unsigned)ix < (unsigned)W;
    const h16* pa = in + (((size_t)b * H + iy) * W + ix) * Cin + 16 * khalf;
    const h16* wt0 = w0 + t * Cin;
    const h16* wt1 = w1 + t * Cin;
    for (int c0 = 0; c0 < Cin; c0 += 32) {
      AFrag a;
      a.h[0] = inb ? *(const v8h*)(pa + c0)      : z;
      a.h[1] = inb ? *(const v8h*)(pa + c0 + 8)  : z;
      v16h b0 = *(const v16h*)(wt0 + c0);
      v16h b1 = *(const v16h*)(wt1 + c0);
      acc0 = WMMA_F16(a.v, b0, acc0);
      acc1 = WMMA_F16(a.v, b1, acc1);
    }
  }
  bn_epilogue(acc0, acc1, scale, shift, nullptr, out,
              mtile, khalf, lrow, nbase, CoutTot, coutOff);
}

// ---------------- stem: 7x7 s4 SAME, Cin=3, 224x224 -> 56x56x96 ----------------
// All dims compile-time so /3 and %3 strength-reduce; K=147 padded to 160.

__global__ __launch_bounds__(256) void stem_bn_wmma(
    const h16* __restrict__ in, const h16* __restrict__ wT,
    const float* __restrict__ scale, const float* __restrict__ shift,
    const float* __restrict__ bias, h16* __restrict__ out)
{
  constexpr int H = 224, W = 224, Cin = 3, KW = 7;
  constexpr int HO = 56, WO = 56, STRIDE = 4, PAD = 1;
  constexpr int Ktot = 147, Kpad = 160, CoutTot = 96;

  const int lane  = threadIdx.x & 31;
  const int wave  = threadIdx.x >> 5;
  const int lrow  = lane & 15;
  const int khalf = lane >> 4;
  const int mtile = blockIdx.x * 8 + wave;
  const int nbase = blockIdx.y * 32;
  const int m     = mtile * 16 + lrow;        // always < M (see invariant)

  const int b  = m / (HO * WO);
  const int r0 = m % (HO * WO);
  const int oy = r0 / WO;
  const int ox = r0 % WO;
  const int iy0 = oy * STRIDE - PAD;
  const int ix0 = ox * STRIDE - PAD;

  const h16* w0 = wT + (size_t)(nbase + lrow) * Kpad + 16 * khalf;
  const h16* w1 = w0 + (size_t)16 * Kpad;

  v8f acc0 = {};
  v8f acc1 = {};
  for (int k0 = 0; k0 < Kpad; k0 += 32) {
    AFrag a;
#pragma unroll
    for (int e = 0; e < 16; ++e) {
      int kf = e + 16 * khalf;
      int kg = k0 + kf;
      h16 val = (h16)0;
      if (kg < Ktot) {
        int ci = kg % Cin;                 // constant divisor -> magic multiply
        int t  = kg / Cin;
        int kx = t % KW;
        int ky = t / KW;
        int iy = iy0 + ky, ix = ix0 + kx;
        if ((unsigned)iy < (unsigned)H && (unsigned)ix < (unsigned)W)
          val = in[(((size_t)b * H + iy) * W + ix) * Cin + ci];
      }
      a.v[e] = val;
    }
    v16h b0 = *(const v16h*)(w0 + k0);
    v16h b1 = *(const v16h*)(w1 + k0);
    acc0 = WMMA_F16(a.v, b0, acc0);
    acc1 = WMMA_F16(a.v, b1, acc1);
  }
  bn_epilogue(acc0, acc1, scale, shift, bias, out,
              mtile, khalf, lrow, nbase, CoutTot, 0);
}

// ---------------- 2x2 maxpool (f16 NHWC), C % 256 == 0 ----------------

__global__ __launch_bounds__(256) void maxpool2_kernel(
    const h16* __restrict__ in, h16* __restrict__ out, int H, int W, int C) {
  const int x = blockIdx.x, y = blockIdx.y, b = blockIdx.z;
  const int HO = H >> 1, WO = W >> 1;
  const h16* p = in  + (((size_t)b * H + 2 * y) * W + 2 * x) * C;
  h16*       q = out + (((size_t)b * HO + y) * WO + x) * C;
  for (int c = threadIdx.x; c < C; c += 256) {
    float a0 = (float)p[c],                 a1 = (float)p[C + c];
    float a2 = (float)p[(size_t)W * C + c], a3 = (float)p[(size_t)W * C + C + c];
    q[c] = (h16)fmaxf(fmaxf(a0, a1), fmaxf(a2, a3));
  }
}

// ---------------- global-avg-pool -> FC(512x1000) -> softmax ----------------

__global__ __launch_bounds__(256) void head_kernel(
    const h16* __restrict__ in, const float* __restrict__ fcw,
    const float* __restrict__ fcb, float* __restrict__ out,
    int HW, int C, int NCLS) {
  __shared__ float feat[512];
  __shared__ float logits[1000];
  __shared__ float red[256];
  const int b = blockIdx.x, t = threadIdx.x;
  const h16* base = in + (size_t)b * HW * C;

  for (int c = t; c < C; c += 256) {
    float s = 0.f;
    for (int p = 0; p < HW; ++p) s += (float)base[(size_t)p * C + c];
    feat[c] = s / (float)HW;
  }
  __syncthreads();

  for (int j = t; j < NCLS; j += 256) {
    float s = fcb[j];
    for (int c = 0; c < C; ++c) s += feat[c] * fcw[(size_t)c * NCLS + j];
    logits[j] = s;
  }
  __syncthreads();

  float mx = -3.0e38f;
  for (int j = t; j < NCLS; j += 256) mx = fmaxf(mx, logits[j]);
  red[t] = mx; __syncthreads();
  for (int s2 = 128; s2 > 0; s2 >>= 1) {
    if (t < s2) red[t] = fmaxf(red[t], red[t + s2]);
    __syncthreads();
  }
  mx = red[0]; __syncthreads();

  float sm = 0.f;
  for (int j = t; j < NCLS; j += 256) {
    float e = __expf(logits[j] - mx);
    logits[j] = e;
    sm += e;
  }
  red[t] = sm; __syncthreads();
  for (int s2 = 128; s2 > 0; s2 >>= 1) {
    if (t < s2) red[t] += red[t + s2];
    __syncthreads();
  }
  float inv = 1.0f / red[0];
  for (int j = t; j < NCLS; j += 256) out[(size_t)b * NCLS + j] = logits[j] * inv;
}

// ---------------- host orchestration ----------------

extern "C" void kernel_launch(void* const* d_in, const int* in_sizes, int n_in,
                              void* d_out, int out_size, void* d_ws, size_t ws_size,
                              hipStream_t stream) {
  (void)in_sizes; (void)n_in; (void)out_size; (void)ws_size;
  const int NB = 32;
  static const int FS[8] = {32, 32, 32, 32, 64, 64, 64, 64};
  static const int FE[8] = {64, 64, 128, 128, 192, 192, 256, 256};

  struct CW { int widx, gidx, kh, kw, cin, cout, bin; };
  CW cw[25];
  cw[0] = {1, 3, 7, 7, 3, 96, 0};              // stem: real-valued weights + bias
  {
    int cin = 96, nc = 1;
    for (int i = 0; i < 8; ++i) {
      int base = 7 + 15 * i;
      cw[nc++] = {base,      base + 1,  1, 1, cin,   FS[i], 1};  // squeeze
      cw[nc++] = {base + 5,  base + 6,  1, 1, FS[i], FE[i], 1};  // expand1x1
      cw[nc++] = {base + 10, base + 11, 3, 3, FS[i], FE[i], 1};  // expand3x3
      cin = 2 * FE[i];
    }
  }

  char* wsb = (char*)d_ws;
  size_t off = 0;
  auto alloc = [&](size_t bytes) -> size_t {
    size_t o = off;
    off = (o + bytes + 255) & ~(size_t)255;
    return o;
  };

  size_t wOff[25], scOff[25], shOff[25];
  int    kpad[25];
  for (int i = 0; i < 25; ++i) {
    int K = cw[i].kh * cw[i].kw * cw[i].cin;
    kpad[i] = (K + 31) & ~31;
    wOff[i]  = alloc((size_t)cw[i].cout * kpad[i] * sizeof(h16));
    scOff[i] = alloc((size_t)cw[i].cout * sizeof(float));
    shOff[i] = alloc((size_t)cw[i].cout * sizeof(float));
  }
  const size_t bufElems = (size_t)NB * 56 * 56 * 256;        // largest activation
  size_t offA = alloc(bufElems * sizeof(h16));
  size_t offB = alloc(bufElems * sizeof(h16));
  size_t offS = alloc((size_t)NB * 56 * 56 * 32 * sizeof(h16));
  h16* bufA = (h16*)(wsb + offA);
  h16* bufB = (h16*)(wsb + offB);
  h16* bufS = (h16*)(wsb + offS);

  // weight transpose/binarize + BN folding
  for (int i = 0; i < 25; ++i) {
    int K = cw[i].kh * cw[i].kw * cw[i].cin;
    long n = (long)cw[i].cout * kpad[i];
    int blk = (int)((n + 255) / 256);
    if (blk > 4096) blk = 4096;
    prep_wT_kernel<<<blk, 256, 0, stream>>>((const float*)d_in[cw[i].widx],
                                            (h16*)(wsb + wOff[i]),
                                            K, kpad[i], cw[i].cout, cw[i].bin);
    fold_bn_kernel<<<(cw[i].cout + 255) / 256, 256, 0, stream>>>(
        (const float*)d_in[cw[i].gidx],     (const float*)d_in[cw[i].gidx + 1],
        (const float*)d_in[cw[i].gidx + 2], (const float*)d_in[cw[i].gidx + 3],
        (float*)(wsb + scOff[i]), (float*)(wsb + shOff[i]), cw[i].cout);
  }

  // x f32 -> f16 (staged in bufB; stem writes bufA so no conflict)
  {
    long nx = (long)NB * 224 * 224 * 3;
    cast_f32_h_kernel<<<(int)((nx + 255) / 256), 256, 0, stream>>>(
        (const float*)d_in[0], bufB, nx);
  }

  // stem
  {
    int M = NB * 56 * 56;
    dim3 grid(M / 128, 96 / 32);
    stem_bn_wmma<<<grid, 256, 0, stream>>>(
        bufB, (const h16*)(wsb + wOff[0]),
        (const float*)(wsb + scOff[0]), (const float*)(wsb + shOff[0]),
        (const float*)d_in[2], bufA);
  }

  h16* cur = bufA;
  h16* nxt = bufB;
  int Hc = 56, Wc = 56;
  int ci = 1;
  for (int i = 0; i < 8; ++i) {
    int e1 = FE[i], e3 = FE[i];
    int cs = ci, ce1 = ci + 1, ce3 = ci + 2;
    ci += 3;
    int M = NB * Hc * Wc;
    dim3 gs(M / 128, FS[i] / 32);
    gemm_bn_wmma<<<gs, 256, 0, stream>>>(                     // squeeze 1x1
        cur, (const h16*)(wsb + wOff[cs]),
        (const float*)(wsb + scOff[cs]), (const float*)(wsb + shOff[cs]),
        bufS, cw[cs].cin, FS[i], 0);
    dim3 ge(M / 128, e1 / 32);
    gemm_bn_wmma<<<ge, 256, 0, stream>>>(                     // expand 1x1
        bufS, (const h16*)(wsb + wOff[ce1]),
        (const float*)(wsb + scOff[ce1]), (const float*)(wsb + shOff[ce1]),
        nxt, FS[i], e1 + e3, 0);
    dim3 g3(M / 128, e3 / 32);
    conv3_bn_wmma<<<g3, 256, 0, stream>>>(                    // expand 3x3
        bufS, (const h16*)(wsb + wOff[ce3]),
        (const float*)(wsb + scOff[ce3]), (const float*)(wsb + shOff[ce3]),
        nxt, Hc, Wc, FS[i], e1 + e3, e1);
    { h16* t = cur; cur = nxt; nxt = t; }
    int Cc = e1 + e3;
    if (i == 2 || i == 6) {
      dim3 gp(Wc / 2, Hc / 2, NB);
      maxpool2_kernel<<<gp, 256, 0, stream>>>(cur, nxt, Hc, Wc, Cc);
      Hc /= 2; Wc /= 2;
      { h16* t = cur; cur = nxt; nxt = t; }
    }
  }

  // global avg pool (14x14x512) -> dense 1000 -> softmax, f32 out
  head_kernel<<<NB, 256, 0, stream>>>(cur, (const float*)d_in[127],
                                      (const float*)d_in[128], (float*)d_out,
                                      Hc * Wc, 512, 1000);
}
